// Decoder_32418413150694
// MI455X (gfx1250) — matrix operand order repaired and checked by reading, not compiled, not run on hardware
//
#include <hip/hip_runtime.h>
#include <cstdint>
#include <cstddef>

// ---------------------------------------------------------------------------
// Problem constants (bs=64, F=1024, n=384, H=P=1024, Kconv=7, V=9)
// ---------------------------------------------------------------------------
#define BS    64
#define FDIM  1024
#define NSEQ  384
#define LSEQ  385      // n + 1 (left zero pad column)
#define HDIM  1024
#define PDIM  1024
#define H3    (3*HDIM)
#define VCLS  9

typedef __attribute__((ext_vector_type(16))) __bf16 v16bf;
typedef __attribute__((ext_vector_type(8)))  __bf16 v8bf;
typedef __attribute__((ext_vector_type(8)))  float  v8f;

// ---------------------------------------------------------------------------
// WMMA core: D = A(16x32 bf16) x B(32x16 bf16) + C(16x16 f32)
// ---------------------------------------------------------------------------
__device__ __forceinline__ v8f wmma_bf16(v16bf a, v16bf b, v8f c) {
#if defined(__gfx1250__)
  return __builtin_amdgcn_wmma_f32_16x16x32_bf16(false, a, false, b, (short)0, c,
                                                 false, false);
#else
  return c;  // host / non-device pass
#endif
}

// A tile 16(M)x32(K), bf16 row-major source. rowp = &A[m_lane][k0] per lane.
// Lane layout (ISA 7.12.2): lanes0-15 K 0-7/16-23, lanes16-31 K 8-15/24-31.
__device__ __forceinline__ v16bf load_a_rowp(const __bf16* rowp) {
  int half = (threadIdx.x >> 4) & 1;
  const __bf16* p = rowp + half * 8;
  v8bf lo = *(const v8bf*)(p);        // K half*8 .. +7   (16B)
  v8bf hi = *(const v8bf*)(p + 16);   // K 16+half*8 .. +7 (16B)
  return __builtin_shufflevector(lo, hi, 0, 1, 2, 3, 4, 5, 6, 7,
                                 8, 9, 10, 11, 12, 13, 14, 15);
}

// B tile 32(K)x16(N) from an N x K bf16 row-major source (B = W^T).
// rowp = &W[n_lane][k0] per lane. lanes0-15 K 0-7/16-23, lanes16-31 K 8-15/24-31.
__device__ __forceinline__ v16bf load_b_rowp(const __bf16* rowp) {
  int kc = ((threadIdx.x >> 4) & 1) * 8;
  const __bf16* p = rowp + kc;
  v8bf lo = *(const v8bf*)(p);        // 16B
  v8bf hi = *(const v8bf*)(p + 16);   // 16B
  return __builtin_shufflevector(lo, hi, 0, 1, 2, 3, 4, 5, 6, 7,
                                 8, 9, 10, 11, 12, 13, 14, 15);
}

// ---------------------------------------------------------------------------
// feats (b, f, l) f32  ->  featsT (b, l+1, f) bf16   (row 0 zeroed separately)
// LDS-tiled 32x32 transpose; coalesced on both sides.
// ---------------------------------------------------------------------------
__global__ __launch_bounds__(256)
void transpose_kernel(const float* __restrict__ feats, __bf16* __restrict__ featsT) {
  __shared__ float tile[32][33];
  int b = blockIdx.z, f0 = blockIdx.y * 32, l0 = blockIdx.x * 32;
  int tx = threadIdx.x, ty = threadIdx.y;  // 32 x 8
  for (int i = ty; i < 32; i += 8)
    tile[i][tx] = feats[((size_t)b * FDIM + f0 + i) * NSEQ + l0 + tx];
  __syncthreads();
  for (int i = ty; i < 32; i += 8)
    featsT[((size_t)b * LSEQ + 1 + l0 + i) * FDIM + f0 + tx] = (__bf16)tile[tx][i];
}

__global__ void zrow_kernel(__bf16* __restrict__ featsT) {
  int i = blockIdx.x * blockDim.x + threadIdx.x;  // BS*FDIM threads
  if (i >= BS * FDIM) return;
  int b = i >> 10, f = i & 1023;
  featsT[((size_t)b * LSEQ) * FDIM + f] = (__bf16)0.f;
}

// Generic f32 (ld = ldsrc) -> packed bf16 (rows x cols, ld = cols)
__global__ void wcvt_kernel(const float* __restrict__ src, int ldsrc,
                            __bf16* __restrict__ dst, int rows, int cols) {
  size_t i = (size_t)blockIdx.x * blockDim.x + threadIdx.x;
  if (i >= (size_t)rows * cols) return;
  size_t r = i / cols, c = i % cols;
  dst[i] = (__bf16)src[r * ldsrc + c];
}

// ---------------------------------------------------------------------------
// Precompute GEMM:  Y[b, l, o] = bias[o] + sum_f W[o, f] * featsT[b, l, f]
// All operands bf16; one wave per 16(o) x 16(l) tile; K = FDIM.
// ---------------------------------------------------------------------------
__global__ __launch_bounds__(256)
void pregemm_kernel(const __bf16* __restrict__ Wbf,
                    const __bf16* __restrict__ featsT,
                    int O,
                    const float* __restrict__ bias,
                    const float* __restrict__ fmask,  // optional mask_p multiply
                    __bf16* __restrict__ Y) {
  int wid  = (blockIdx.x * blockDim.x + threadIdx.x) >> 5;
  int lane = threadIdx.x & 31;
  int otiles = O >> 4;
  const int ltiles = (LSEQ + 15) >> 4;  // 25
  int ot = wid % otiles;
  int lt = (wid / otiles) % ltiles;
  int b  = wid / (otiles * ltiles);
  if (b >= BS) return;
  int o0 = ot * 16, l0 = lt * 16;

  int n = lane & 15;
  int lcol = l0 + n;
  int lrow = (lcol < LSEQ) ? lcol : 0;  // clamp OOB lanes to zero row (unused)
  const __bf16* arow = Wbf + (size_t)(o0 + n) * FDIM;       // n == m for A lanes
  const __bf16* brow = featsT + ((size_t)b * LSEQ + lrow) * FDIM;

  v8f c = {0.f, 0.f, 0.f, 0.f, 0.f, 0.f, 0.f, 0.f};
  for (int k0 = 0; k0 < FDIM; k0 += 32) {
    v16bf a  = load_a_rowp(arow + k0);
    v16bf bm = load_b_rowp(brow + k0);
    c = wmma_bf16(a, bm, c);
  }

  if (lcol >= LSEQ) return;
  float mval = 1.f;
  if (fmask) mval = (lcol == 0) ? 1.f : fmask[(size_t)b * NSEQ + (lcol - 1)];
  int half = lane >> 4;
#pragma unroll
  for (int v = 0; v < 8; ++v) {
    int o = o0 + v + 8 * half;
    float val = c[v] + (bias ? bias[o] : 0.f);
    Y[((size_t)b * LSEQ + lcol) * O + o] = (__bf16)(val * mval);
  }
}

// ---------------------------------------------------------------------------
// Step GEMM:  out[b, n] = sum_k A[b, k] * W[n, k]  (+ addC[b][n] bf16) (+ bias[n])
// A: BS x 1024 bf16 packed.  W: N x 1024 bf16 packed.
// ---------------------------------------------------------------------------
__global__ __launch_bounds__(256)
void step_gemm_kernel(const __bf16* __restrict__ A,
                      const __bf16* __restrict__ W,
                      int N,
                      const __bf16* __restrict__ addC, size_t addC_bstride,
                      const float* __restrict__ bias,
                      float* __restrict__ out) {
  int wid  = (blockIdx.x * blockDim.x + threadIdx.x) >> 5;
  int lane = threadIdx.x & 31;
  int ntiles = N >> 4;
  if (wid >= 4 * ntiles) return;  // BS/16 == 4 m-tiles
  int nt = wid % ntiles;
  int mt = wid / ntiles;
  int m0 = mt * 16, n0 = nt * 16;

  int mn = lane & 15;
  const __bf16* arow = A + (size_t)(m0 + mn) * FDIM;
  const __bf16* brow = W + (size_t)(n0 + mn) * FDIM;

  v8f c = {0.f, 0.f, 0.f, 0.f, 0.f, 0.f, 0.f, 0.f};
  for (int k0 = 0; k0 < FDIM; k0 += 32) {
    v16bf a  = load_a_rowp(arow + k0);
    v16bf bm = load_b_rowp(brow + k0);
    c = wmma_bf16(a, bm, c);
  }

  int half = lane >> 4;
  int col = n0 + mn;
#pragma unroll
  for (int v = 0; v < 8; ++v) {
    int m = m0 + v + 8 * half;
    float val = c[v];
    if (addC) val += (float)addC[(size_t)m * addC_bstride + col];
    if (bias) val += bias[col];
    out[(size_t)m * N + col] = val;
  }
}

// ---------------------------------------------------------------------------
// GRU gate combine: hout = (1-z)*g + z*hin   (gi gathered from bf16 rows)
// row = idxptr ? idxptr[b] : trow
// ---------------------------------------------------------------------------
__global__ __launch_bounds__(256)
void gru_combine_kernel(const __bf16* __restrict__ GI,
                        const int* __restrict__ idxptr, int trow,
                        const float* __restrict__ b_ih,
                        const float* __restrict__ b_hh,
                        const float* __restrict__ gh,
                        const float* __restrict__ hin,
                        float* __restrict__ hout) {
  int i = blockIdx.x * blockDim.x + threadIdx.x;
  if (i >= BS * HDIM) return;
  int b = i >> 10, j = i & 1023;
  int row = idxptr ? idxptr[b] : trow;
  const __bf16* gi  = GI + ((size_t)b * LSEQ + row) * H3;
  const float*  ghb = gh + (size_t)b * H3;
  float ir = (float)gi[j]            + b_ih[j];
  float iz = (float)gi[HDIM + j]     + b_ih[HDIM + j];
  float ig = (float)gi[2 * HDIM + j] + b_ih[2 * HDIM + j];
  float hr = ghb[j]            + b_hh[j];
  float hz = ghb[HDIM + j]     + b_hh[HDIM + j];
  float hg = ghb[2 * HDIM + j] + b_hh[2 * HDIM + j];
  float r = 1.f / (1.f + __expf(-(ir + hr)));
  float z = 1.f / (1.f + __expf(-(iz + hz)));
  float g = tanhf(ig + r * hg);
  hout[i] = (1.f - z) * g + z * hin[i];
}

// ---------------------------------------------------------------------------
// Attention energy: logit[b,l] = blog + sum_p wlog[p] * tanh(PF[b,l,p] + q[b,p]
//        + bcov[p]+bcum[p] + sum_k Wcov[p,k]*saw[l+k-3] + Wcum[p,k]*csaw[l+k-3])
// One wave per (b,l); lanes stripe over p; shuffle reduction.
// ---------------------------------------------------------------------------
__global__ __launch_bounds__(256)
void energy_kernel(const __bf16* __restrict__ PF, const float* __restrict__ q,
                   const float* __restrict__ saw, const float* __restrict__ csaw,
                   const float* __restrict__ Wcov, const float* __restrict__ bcov,
                   const float* __restrict__ Wcum, const float* __restrict__ bcum,
                   const float* __restrict__ Wlog, const float* __restrict__ blog,
                   float* __restrict__ logit) {
  int wid  = (blockIdx.x * blockDim.x + threadIdx.x) >> 5;
  int lane = threadIdx.x & 31;
  int l = wid % LSEQ;
  int b = wid / LSEQ;
  if (b >= BS) return;

  float s[7], cs[7];
#pragma unroll
  for (int k = 0; k < 7; ++k) {
    int ll = l - 3 + k;
    bool ok = (ll >= 0 && ll < LSEQ);
    s[k]  = ok ? saw[(size_t)b * LSEQ + ll]  : 0.f;
    cs[k] = ok ? csaw[(size_t)b * LSEQ + ll] : 0.f;
  }
  const __bf16* pf = PF + ((size_t)b * LSEQ + l) * PDIM;
  const float*  qb = q + (size_t)b * PDIM;

  float acc = 0.f;
  for (int p = lane; p < PDIM; p += 32) {
    if (p + 128 < PDIM) __builtin_prefetch(pf + p + 128, 0, 3);
    float e = (float)pf[p] + qb[p] + bcov[p] + bcum[p];
    const float* wv = Wcov + (size_t)p * 7;
    const float* wu = Wcum + (size_t)p * 7;
#pragma unroll
    for (int k = 0; k < 7; ++k) e += wv[k] * s[k] + wu[k] * cs[k];
    acc += Wlog[p] * tanhf(e);
  }
#pragma unroll
  for (int off = 16; off; off >>= 1) acc += __shfl_xor(acc, off, 32);
  if (lane == 0) logit[(size_t)b * LSEQ + l] = acc + blog[0];
}

// ---------------------------------------------------------------------------
// Masked softmax + saw/csaw update + argmax(sel) + ma_logits output.
// ---------------------------------------------------------------------------
__global__ __launch_bounds__(128)
void softmax_argmax_kernel(const float* __restrict__ logit,
                           const float* __restrict__ fmask,
                           float* __restrict__ saw, float* __restrict__ csaw,
                           int* __restrict__ idxbuf,
                           float* __restrict__ out_ma, int t) {
  __shared__ float red[128];
  __shared__ int   redi[128];
  int b = blockIdx.x, tid = threadIdx.x;
  const float* lg = logit + (size_t)b * LSEQ;

  float mx = -3.4e38f;
  for (int l = tid; l < LSEQ; l += 128) {
    float mp = (l == 0) ? 1.f : fmask[(size_t)b * NSEQ + l - 1];
    float mt = (l <= t) ? mp : 0.f;
    mx = fmaxf(mx, lg[l] - (1.f - mt) * 1e8f);
  }
  red[tid] = mx; __syncthreads();
  for (int s2 = 64; s2; s2 >>= 1) {
    if (tid < s2) red[tid] = fmaxf(red[tid], red[tid + s2]);
    __syncthreads();
  }
  mx = red[0]; __syncthreads();

  float sum = 0.f;
  for (int l = tid; l < LSEQ; l += 128) {
    float mp = (l == 0) ? 1.f : fmask[(size_t)b * NSEQ + l - 1];
    float mt = (l <= t) ? mp : 0.f;
    sum += __expf(lg[l] - (1.f - mt) * 1e8f - mx);
  }
  red[tid] = sum; __syncthreads();
  for (int s2 = 64; s2; s2 >>= 1) {
    if (tid < s2) red[tid] += red[tid + s2];
    __syncthreads();
  }
  float inv = 1.f / red[0]; __syncthreads();

  for (int l = tid; l < LSEQ; l += 128) {
    float mp = (l == 0) ? 1.f : fmask[(size_t)b * NSEQ + l - 1];
    float mt = (l <= t) ? mp : 0.f;
    float att = __expf(lg[l] - (1.f - mt) * 1e8f - mx) * inv;
    saw[(size_t)b * LSEQ + l] = att;
    csaw[(size_t)b * LSEQ + l] += att;
  }

  float best = -3.4e38f;
  int   bi   = 0x7fffffff;
  for (int l = tid; l < LSEQ; l += 128) {
    float sv = (l <= t) ? lg[l] : -1e9f;
    if (sv > best) { best = sv; bi = l; }
  }
  red[tid] = best; redi[tid] = bi; __syncthreads();
  for (int s2 = 64; s2; s2 >>= 1) {
    if (tid < s2) {
      if (red[tid + s2] > red[tid] ||
          (red[tid + s2] == red[tid] && redi[tid + s2] < redi[tid])) {
        red[tid] = red[tid + s2]; redi[tid] = redi[tid + s2];
      }
    }
    __syncthreads();
  }
  if (tid == 0) idxbuf[b] = redi[0];

  for (int l = tid; l < NSEQ; l += 128)
    out_ma[((size_t)b * NSEQ + t) * NSEQ + l] = lg[l];
}

// ---------------------------------------------------------------------------
// Classifier head:  pred = argmax_v( b2[v] + sum_h W2[v,h]*tanh(C1C+C1P+G) )
// ---------------------------------------------------------------------------
__global__ __launch_bounds__(256)
void cls_finalize_kernel(const __bf16* __restrict__ C1C,
                         const __bf16* __restrict__ C1P,
                         const float* __restrict__ G,
                         const float* __restrict__ W2,
                         const float* __restrict__ b2,
                         const int* __restrict__ idxbuf,
                         float* __restrict__ out_pred, int t) {
  __shared__ float red[VCLS][256];
  int b = blockIdx.x, tid = threadIdx.x;
  int row = idxbuf[b];
  const __bf16* cc = C1C + ((size_t)b * LSEQ + t + 1) * HDIM;
  const __bf16* cp = C1P + ((size_t)b * LSEQ + row) * HDIM;
  const float*  gb = G + (size_t)b * HDIM;

  float acc[VCLS];
#pragma unroll
  for (int v = 0; v < VCLS; ++v) acc[v] = 0.f;
  for (int h = tid; h < HDIM; h += 256) {
    float hm = tanhf((float)cc[h] + (float)cp[h] + gb[h]);
#pragma unroll
    for (int v = 0; v < VCLS; ++v) acc[v] += hm * W2[(size_t)v * HDIM + h];
  }
#pragma unroll
  for (int v = 0; v < VCLS; ++v) red[v][tid] = acc[v];
  __syncthreads();
  for (int s2 = 128; s2; s2 >>= 1) {
    if (tid < s2)
#pragma unroll
      for (int v = 0; v < VCLS; ++v) red[v][tid] += red[v][tid + s2];
    __syncthreads();
  }
  if (tid == 0) {
    float best = -3.4e38f; int bi = 0;
    for (int v = 0; v < VCLS; ++v) {
      float val = red[v][0] + b2[v];
      if (val > best) { best = val; bi = v; }
    }
    out_pred[(size_t)b * NSEQ + t] = (float)bi;
  }
}

// ---------------------------------------------------------------------------
// Init helpers
// ---------------------------------------------------------------------------
__global__ void zero_saw_kernel(float* saw, float* csaw) {
  int i = blockIdx.x * blockDim.x + threadIdx.x;
  if (i < BS * LSEQ) { saw[i] = 0.f; csaw[i] = 0.f; }
}

__global__ void nm_kernel(const float* __restrict__ fmask, float* __restrict__ nm) {
  int b = blockIdx.x, lane = threadIdx.x;
  float s = 0.f;
  for (int l = lane; l < NSEQ; l += 32) s += fmask[(size_t)b * NSEQ + l];
#pragma unroll
  for (int off = 16; off; off >>= 1) s += __shfl_xor(s, off, 32);
  if (lane == 0) nm[b] = s + 1.f;  // + pad column
}

// fbar[b,f] = (sum_l mask * feats_p[b,f,l]) / nm[b]
__global__ __launch_bounds__(256)
void fbar_kernel(const float* __restrict__ feats, const float* __restrict__ fmask,
                 const float* __restrict__ nm, float* __restrict__ fbar) {
  int wid  = (blockIdx.x * blockDim.x + threadIdx.x) >> 5;
  int lane = threadIdx.x & 31;
  int f = wid & 1023, b = wid >> 10;
  if (b >= BS) return;
  const float* row = feats + ((size_t)b * FDIM + f) * NSEQ;
  float s = 0.f;
  for (int l = lane; l < NSEQ; l += 32) s += row[l] * fmask[(size_t)b * NSEQ + l];
#pragma unroll
  for (int off = 16; off; off >>= 1) s += __shfl_xor(s, off, 32);
  if (lane == 0) fbar[(size_t)b * FDIM + f] = s / nm[b];
}

// ---------------------------------------------------------------------------
// Host driver
// ---------------------------------------------------------------------------
extern "C" void kernel_launch(void* const* d_in, const int* in_sizes, int n_in,
                              void* d_out, int out_size, void* d_ws, size_t ws_size,
                              hipStream_t stream) {
  (void)in_sizes; (void)n_in; (void)out_size; (void)ws_size;

  const float* feats      = (const float*)d_in[0];
  const float* feats_mask = (const float*)d_in[1];
  const float* W_fp  = (const float*)d_in[2];
  const float* b_fp  = (const float*)d_in[3];
  const float* W_si  = (const float*)d_in[4];
  const float* b_si  = (const float*)d_in[5];
  const float* W_ih1 = (const float*)d_in[6];
  const float* W_hh1 = (const float*)d_in[7];
  const float* b_ih1 = (const float*)d_in[8];
  const float* b_hh1 = (const float*)d_in[9];
  const float* W_ih2 = (const float*)d_in[10];
  const float* W_hh2 = (const float*)d_in[11];
  const float* b_ih2 = (const float*)d_in[12];
  const float* b_hh2 = (const float*)d_in[13];
  const float* Wq    = (const float*)d_in[14];
  const float* bq    = (const float*)d_in[15];
  const float* Wcov  = (const float*)d_in[16];
  const float* bcov  = (const float*)d_in[17];
  const float* Wcum  = (const float*)d_in[18];
  const float* bcum  = (const float*)d_in[19];
  const float* Wlog  = (const float*)d_in[20];
  const float* blog  = (const float*)d_in[21];
  const float* W1    = (const float*)d_in[22];
  const float* b1    = (const float*)d_in[23];
  const float* W2    = (const float*)d_in[24];
  const float* b2    = (const float*)d_in[25];

  float* outf     = (float*)d_out;
  float* out_pred = outf;                       // (64, 384) argmax as float
  float* out_ma   = outf + (size_t)BS * NSEQ;   // (64, 384, 384)

  // --- scratch layout (bump allocator, 256B aligned) ---
  size_t off = 0;
  char*  base = (char*)d_ws;
  auto alloc = [&](size_t bytes) -> void* {
    off = (off + 255) & ~(size_t)255;
    void* p = base + off;
    off += bytes;
    return p;
  };
  // transposed bf16 feats (pad row 0)
  __bf16* featsT = (__bf16*)alloc((size_t)BS * LSEQ * FDIM * 2);
  // bf16 weights, packed N x 1024
  __bf16* Wfp_bf  = (__bf16*)alloc((size_t)PDIM * FDIM * 2);
  __bf16* Wsi_bf  = (__bf16*)alloc((size_t)HDIM * FDIM * 2);
  __bf16* Wih1_bf = (__bf16*)alloc((size_t)H3 * FDIM * 2);
  __bf16* Wih2_bf = (__bf16*)alloc((size_t)H3 * FDIM * 2);
  __bf16* Whh1_bf = (__bf16*)alloc((size_t)H3 * HDIM * 2);
  __bf16* Whh2_bf = (__bf16*)alloc((size_t)H3 * HDIM * 2);
  __bf16* Wqc_bf  = (__bf16*)alloc((size_t)PDIM * FDIM * 2);
  __bf16* Wqh_bf  = (__bf16*)alloc((size_t)PDIM * HDIM * 2);
  __bf16* W1c_bf  = (__bf16*)alloc((size_t)HDIM * FDIM * 2);
  __bf16* W1p_bf  = (__bf16*)alloc((size_t)HDIM * FDIM * 2);
  __bf16* W1h_bf  = (__bf16*)alloc((size_t)HDIM * HDIM * 2);
  // precomputed activations (all LSEQ rows; context arrays use row t+1)
  __bf16* GI1 = (__bf16*)alloc((size_t)BS * LSEQ * H3 * 2);
  __bf16* GI2 = (__bf16*)alloc((size_t)BS * LSEQ * H3 * 2);
  __bf16* PF  = (__bf16*)alloc((size_t)BS * LSEQ * PDIM * 2);
  __bf16* QC  = (__bf16*)alloc((size_t)BS * LSEQ * PDIM * 2);
  __bf16* C1C = (__bf16*)alloc((size_t)BS * LSEQ * HDIM * 2);
  __bf16* C1P = (__bf16*)alloc((size_t)BS * LSEQ * HDIM * 2);
  // per-step state
  float*  h     = (float*)alloc((size_t)BS * HDIM * 4);
  float*  ns    = (float*)alloc((size_t)BS * HDIM * 4);
  __bf16* hbf   = (__bf16*)alloc((size_t)BS * HDIM * 2);
  __bf16* nsbf  = (__bf16*)alloc((size_t)BS * HDIM * 2);
  float*  gh1   = (float*)alloc((size_t)BS * H3 * 4);
  float*  gh2   = (float*)alloc((size_t)BS * H3 * 4);
  float*  q     = (float*)alloc((size_t)BS * PDIM * 4);
  float*  G     = (float*)alloc((size_t)BS * HDIM * 4);
  float*  fbar  = (float*)alloc((size_t)BS * FDIM * 4);
  float*  logit = (float*)alloc((size_t)BS * LSEQ * 4);
  float*  saw   = (float*)alloc((size_t)BS * LSEQ * 4);
  float*  csaw  = (float*)alloc((size_t)BS * LSEQ * 4);
  float*  nm    = (float*)alloc((size_t)BS * 4);
  int*    idx   = (int*)alloc((size_t)BS * 4);

  auto blocksForWaves = [](long long waves) -> int {
    return (int)((waves * 32 + 255) / 256);
  };
  auto wcvt = [&](const float* src, int ldsrc, __bf16* dst, int rows, int cols) {
    long long nelem = (long long)rows * cols;
    wcvt_kernel<<<(int)((nelem + 255) / 256), 256, 0, stream>>>(src, ldsrc, dst,
                                                                rows, cols);
  };

  // --- layout conversions (once) ---
  transpose_kernel<<<dim3(NSEQ / 32, FDIM / 32, BS), dim3(32, 8), 0, stream>>>(
      feats, featsT);
  zrow_kernel<<<(BS * FDIM + 255) / 256, 256, 0, stream>>>(featsT);
  wcvt(W_fp,              FDIM,            Wfp_bf,  PDIM, FDIM);
  wcvt(W_si,              FDIM,            Wsi_bf,  HDIM, FDIM);
  wcvt(W_ih1,             FDIM,            Wih1_bf, H3,   FDIM);
  wcvt(W_ih2,             FDIM,            Wih2_bf, H3,   FDIM);
  wcvt(W_hh1,             HDIM,            Whh1_bf, H3,   HDIM);
  wcvt(W_hh2,             HDIM,            Whh2_bf, H3,   HDIM);
  wcvt(Wq,                FDIM + HDIM,     Wqc_bf,  PDIM, FDIM);
  wcvt(Wq + FDIM,         FDIM + HDIM,     Wqh_bf,  PDIM, HDIM);
  wcvt(W1,                2 * FDIM + HDIM, W1c_bf,  HDIM, FDIM);
  wcvt(W1 + FDIM,         2 * FDIM + HDIM, W1p_bf,  HDIM, FDIM);
  wcvt(W1 + 2 * FDIM,     2 * FDIM + HDIM, W1h_bf,  HDIM, HDIM);

  // --- init state ---
  zero_saw_kernel<<<(BS * LSEQ + 255) / 256, 256, 0, stream>>>(saw, csaw);
  nm_kernel<<<BS, 32, 0, stream>>>(feats_mask, nm);
  fbar_kernel<<<blocksForWaves((long long)BS * FDIM), 256, 0, stream>>>(
      feats, feats_mask, nm, fbar);
  wcvt(fbar, FDIM, nsbf, BS, FDIM);  // reuse nsbf as fbar_bf16
  step_gemm_kernel<<<blocksForWaves(4LL * (HDIM / 16)), 256, 0, stream>>>(
      nsbf, Wsi_bf, HDIM, nullptr, 0, b_si, h);  // state0 -> h

  // --- precompute big GEMMs (WMMA bf16, all over LSEQ rows) ---
  auto pre = [&](const __bf16* Wbf, int O, const float* bias, const float* fm,
                 __bf16* Y) {
    long long waves = (long long)BS * (O / 16) * ((LSEQ + 15) / 16);
    pregemm_kernel<<<blocksForWaves(waves), 256, 0, stream>>>(Wbf, featsT, O,
                                                              bias, fm, Y);
  };
  pre(Wfp_bf,  PDIM, b_fp,    feats_mask, PF);
  pre(Wih1_bf, H3,   nullptr, nullptr,    GI1);
  pre(Wih2_bf, H3,   nullptr, nullptr,    GI2);
  pre(Wqc_bf,  PDIM, bq,      nullptr,    QC);
  pre(W1c_bf,  HDIM, b1,      nullptr,    C1C);
  pre(W1p_bf,  HDIM, nullptr, nullptr,    C1P);

  auto sgemm = [&](const __bf16* A, const __bf16* W, int N, const __bf16* addC,
                   size_t addCbs, const float* bias, float* out) {
    step_gemm_kernel<<<blocksForWaves(4LL * (N / 16)), 256, 0, stream>>>(
        A, W, N, addC, addCbs, bias, out);
  };

  // initial activation conversion
  wcvt(h, HDIM, hbf, BS, HDIM);

  // --- sequential scan over 384 timesteps ---
  for (int t = 0; t < NSEQ; ++t) {
    // GRU1: gh1 = h @ W_hh1^T ; ns = gru(GI1[t+1], gh1, h)
    sgemm(hbf, Whh1_bf, H3, nullptr, 0, nullptr, gh1);
    gru_combine_kernel<<<(BS * HDIM + 255) / 256, 256, 0, stream>>>(
        GI1, nullptr, t + 1, b_ih1, b_hh1, gh1, h, ns);
    wcvt(ns, HDIM, nsbf, BS, HDIM);

    // q = QC[t+1] + ns @ Wq[:, F:]^T   (bq folded into QC)
    sgemm(nsbf, Wqh_bf, PDIM, QC + (size_t)(t + 1) * PDIM,
          (size_t)LSEQ * PDIM, nullptr, q);

    // attention energies + masked softmax/argmax (updates saw, csaw, idx, ma)
    energy_kernel<<<blocksForWaves((long long)BS * LSEQ), 256, 0, stream>>>(
        PF, q, saw, csaw, Wcov, bcov, Wcum, bcum, Wlog, blog, logit);
    softmax_argmax_kernel<<<BS, 128, 0, stream>>>(
        logit, feats_mask, saw, csaw, idx, out_ma, t);

    // GRU2: gh2 = ns @ W_hh2^T ; h(next) = gru(GI2[idx], gh2, ns)
    sgemm(nsbf, Whh2_bf, H3, nullptr, 0, nullptr, gh2);
    gru_combine_kernel<<<(BS * HDIM + 255) / 256, 256, 0, stream>>>(
        GI2, idx, 0, b_ih2, b_hh2, gh2, ns, h);
    wcvt(h, HDIM, hbf, BS, HDIM);

    // classifier: G = ns2 @ W1[:, 2F:]^T ; pred = argmax head
    sgemm(hbf, W1h_bf, HDIM, nullptr, 0, nullptr, G);
    cls_finalize_kernel<<<BS, 256, 0, stream>>>(
        C1C, C1P, G, W2, b2, idx, out_pred, t);
  }
}
